// PrunableMixtralSparseMoeBlockWrapper_87668872446485
// MI455X (gfx1250) — matrix operand order repaired and checked by reading, not compiled, not run on hardware
//
#include <hip/hip_runtime.h>
#include <hip/hip_bf16.h>

// ---------------- problem constants (from reference) ----------------
#define T_TOK   8192          // B*S
#define H_DIM   1024
#define F_DIM   4096
#define N_EXP   8
#define TILE_M  16
#define FC      128           // F-chunk width processed per iteration
#define MAX_TILES (T_TOK / TILE_M)   // 512 (worst case: all tokens on one expert)

typedef __attribute__((ext_vector_type(16))) __bf16  bf16x16;
typedef __attribute__((ext_vector_type(8)))  float   f32x8;

#if defined(__HIP_DEVICE_COMPILE__) && \
    __has_builtin(__builtin_amdgcn_global_load_async_to_lds_b128) && \
    __has_builtin(__builtin_amdgcn_s_wait_asynccnt)
#define USE_ASYNC_LDS 1
// builtin signature (from compiler diagnostic): param0 = global ptr to 16B int vector
typedef int v4i_ __attribute__((__vector_size__(16)));
typedef __attribute__((address_space(1))) v4i_ gv4i;   // global
typedef __attribute__((address_space(3))) v4i_ lv4i;   // LDS
#else
#define USE_ASYNC_LDS 0
#endif

// ---------------- fragment loaders (CDNA5 wave32 WMMA layouts) ----------------

// A-matrix 16x32 bf16 fragment from LDS (row-major tile, leading dim `ld`).
// lanes 0-15: M=lane, K = k0 + {0..15}; lanes 16-31: M=lane-16, K = k0 + {16..31}
__device__ __forceinline__ bf16x16 load_a_lds(const __bf16* base, int ld, int k0, int lane) {
    const int m  = lane & 15;
    const int kh = (lane >> 4) << 4;          // 0 or 16
    const __bf16* p = base + m * ld + k0 + kh;
    bf16x16 a;
    ((uint4*)&a)[0] = *(const uint4*)(p);       // K = k0+kh+0..7
    ((uint4*)&a)[1] = *(const uint4*)(p + 8);   // K = k0+kh+8..15
    return a;
}

// B-matrix 32x16 fragment, element (k,n) = W[(n0+n)*row_stride + k0 + k].
// lanes 0-15: K=0..15, lanes 16-31: K=16..31.

// f32 weights: convert on the fly (fallback path when workspace is small).
__device__ __forceinline__ bf16x16 load_b(const float* __restrict__ base,
                                          size_t row_stride, int n0, int k0, int lane) {
    const int n  = (lane & 15) + n0;
    const int kh = (lane >> 4) << 4;          // 0 or 16
    const float4* p = (const float4*)(base + (size_t)n * row_stride + (size_t)(k0 + kh));
    bf16x16 b;
#pragma unroll
    for (int j = 0; j < 4; ++j) {
        float4 q = p[j];
        b[4*j+0] = (__bf16)q.x; b[4*j+1] = (__bf16)q.y;
        b[4*j+2] = (__bf16)q.z; b[4*j+3] = (__bf16)q.w;
    }
    return b;
}

// bf16 weights (pre-converted): two contiguous 16B loads, no VALU conversion.
__device__ __forceinline__ bf16x16 load_b(const __bf16* __restrict__ base,
                                          size_t row_stride, int n0, int k0, int lane) {
    const int n  = (lane & 15) + n0;
    const int kh = (lane >> 4) << 4;          // 0 or 16
    const __bf16* p = base + (size_t)n * row_stride + (size_t)(k0 + kh);
    bf16x16 b;
    ((uint4*)&b)[0] = *(const uint4*)(p);      // K = kh+0..7
    ((uint4*)&b)[1] = *(const uint4*)(p + 8);  // K = kh+8..15
    return b;
}

__device__ __forceinline__ f32x8 wmma_bf16(bf16x16 a, bf16x16 b, f32x8 c) {
    // (neg_a, A, neg_b, B, c_mod, C, reuse_a, reuse_b)
    return __builtin_amdgcn_wmma_f32_16x16x32_bf16(false, a, false, b, (short)0, c, false, false);
}

// ---------------- kernel 1: zero output accumulation region + counters ----------------
__global__ __launch_bounds__(256)
void init_kernel(float* __restrict__ out, int* __restrict__ counts) {
    size_t i = (size_t)blockIdx.x * blockDim.x + threadIdx.x;
    if (i < N_EXP) counts[i] = 0;
    const size_t n = (size_t)T_TOK * H_DIM;
    for (size_t j = i; j < n; j += (size_t)gridDim.x * blockDim.x) out[j] = 0.0f;
}

// ---------------- kernel 2: activations f32 -> bf16 ----------------
__device__ __forceinline__ void cvt4(const float* __restrict__ s, __bf16* __restrict__ d, size_t i) {
    float4 v = *(const float4*)(s + i);
    __bf16 t[4] = { (__bf16)v.x, (__bf16)v.y, (__bf16)v.z, (__bf16)v.w };
    *(uint2*)(d + i) = *(const uint2*)t;
}

__global__ __launch_bounds__(256)
void cvt_x_kernel(const float* __restrict__ x, __bf16* __restrict__ xb) {
    size_t i = ((size_t)blockIdx.x * blockDim.x + threadIdx.x) * 4;
    cvt4(x, xb, i);
}

// ---------------- kernel 2b: expert weights f32 -> bf16 (one pass, 3 tensors) --------
__global__ __launch_bounds__(256)
void cvt_w_kernel(const float* __restrict__ w1, __bf16* __restrict__ w1b,
                  const float* __restrict__ w3, __bf16* __restrict__ w3b,
                  const float* __restrict__ w2, __bf16* __restrict__ w2b) {
    size_t i = ((size_t)blockIdx.x * blockDim.x + threadIdx.x) * 4;
    cvt4(w1, w1b, i);
    cvt4(w3, w3b, i);
    cvt4(w2, w2b, i);
}

// ---------------- kernel 3: router (logits, softmax, top-2, bucket by expert) -------
__global__ __launch_bounds__(256)
void router_kernel(const float* __restrict__ x, const float* __restrict__ gw,
                   float* __restrict__ logits_out, int* __restrict__ counts,
                   int* __restrict__ tix, float* __restrict__ twt) {
    const int t    = blockIdx.x * 8 + (threadIdx.x >> 5);   // one wave32 per token
    const int lane = threadIdx.x & 31;
    const float* xr = x + (size_t)t * H_DIM;

    float acc[N_EXP];
#pragma unroll
    for (int e = 0; e < N_EXP; ++e) acc[e] = 0.0f;

    for (int k = lane; k < H_DIM; k += 32) {
        float xv = xr[k];
#pragma unroll
        for (int e = 0; e < N_EXP; ++e) acc[e] += xv * gw[e * H_DIM + k];
    }
#pragma unroll
    for (int e = 0; e < N_EXP; ++e) {
#pragma unroll
        for (int off = 16; off > 0; off >>= 1) acc[e] += __shfl_xor(acc[e], off, 32);
    }

    if (lane == 0) {
        float mx = acc[0];
#pragma unroll
        for (int e = 1; e < N_EXP; ++e) mx = fmaxf(mx, acc[e]);
        float ex[N_EXP], s = 0.0f;
#pragma unroll
        for (int e = 0; e < N_EXP; ++e) { ex[e] = __expf(acc[e] - mx); s += ex[e]; }

        int i0 = 0;
#pragma unroll
        for (int e = 1; e < N_EXP; ++e) if (ex[e] > ex[i0]) i0 = e;   // stable: first max
        int i1 = (i0 == 0) ? 1 : 0;
#pragma unroll
        for (int e = 0; e < N_EXP; ++e) if (e != i0 && ex[e] > ex[i1]) i1 = e;

        float p0 = ex[i0] / s, p1 = ex[i1] / s;
        float rn = 1.0f / (p0 + p1);

#pragma unroll
        for (int e = 0; e < N_EXP; ++e) logits_out[(size_t)t * N_EXP + e] = acc[e];

        int pos0 = atomicAdd(&counts[i0], 1);
        tix[(size_t)i0 * T_TOK + pos0] = t;  twt[(size_t)i0 * T_TOK + pos0] = p0 * rn;
        int pos1 = atomicAdd(&counts[i1], 1);
        tix[(size_t)i1 * T_TOK + pos1] = t;  twt[(size_t)i1 * T_TOK + pos1] = p1 * rn;
    }
}

// ---------------- kernel 4: fused expert FFN (grouped GEMM, bf16 WMMA) ----------------
// One workgroup = 8 wave32s, one 16-token tile of one expert.
// out[tok] += w * ( silu(x w1^T) * (x w3^T) ) w2^T
// WT = __bf16 (pre-converted weights, preferred) or float (on-the-fly cvt fallback).
template <typename WT>
__global__ __launch_bounds__(256)
void moe_ffn_kernel(const WT* __restrict__ w1, const WT* __restrict__ w3,
                    const WT* __restrict__ w2, const __bf16* __restrict__ xb,
                    const int* __restrict__ counts, const int* __restrict__ tix,
                    const float* __restrict__ twt, float* __restrict__ out) {
    const int e    = blockIdx.y;
    const int tile = blockIdx.x;
    const int cnt  = counts[e];
    if (tile * TILE_M >= cnt) return;     // uniform early-exit (no barrier yet)

    __shared__ __align__(16) __bf16 xs[TILE_M * H_DIM];  // 32 KB: 16 token rows, bf16
    __shared__ __align__(16) __bf16 hs[TILE_M * FC];     //  4 KB: h chunk, bf16
    __shared__ float twl[TILE_M];
    __shared__ int   tokl[TILE_M];

    const int tid  = threadIdx.x;
    const int lane = tid & 31;
    const int wave = tid >> 5;

    if (tid < TILE_M) {
        int r = tile * TILE_M + tid;
        bool ok = r < cnt;
        tokl[tid] = ok ? tix[(size_t)e * T_TOK + r] : 0;
        twl[tid]  = ok ? twt[(size_t)e * T_TOK + r] : 0.0f;
    }
    __syncthreads();

    {   // stage 16 x-rows (bf16) into LDS; zero-fill padded rows
        int row = tid >> 4, sub = tid & 15;               // 16 threads per row
        bool valid = (tile * TILE_M + row) < cnt;
        __bf16* dst = xs + row * H_DIM + sub * 64;
        const __bf16* src = xb + (size_t)tokl[row] * H_DIM + sub * 64;
#if USE_ASYNC_LDS
        if (valid) {
#pragma unroll
            for (int j = 0; j < 8; ++j)
                __builtin_amdgcn_global_load_async_to_lds_b128(
                    (gv4i*)(const_cast<__bf16*>(src) + j * 8),
                    (lv4i*)(dst + j * 8), 0, 0);
        } else {
            uint4 z = make_uint4(0u, 0u, 0u, 0u);
#pragma unroll
            for (int j = 0; j < 8; ++j) ((uint4*)dst)[j] = z;
        }
        __builtin_amdgcn_s_wait_asynccnt(0);
#else
        uint4 z = make_uint4(0u, 0u, 0u, 0u);
#pragma unroll
        for (int j = 0; j < 8; ++j)
            ((uint4*)dst)[j] = valid ? ((const uint4*)src)[j] : z;
#endif
    }
    __syncthreads();

    const WT* w1e = w1 + (size_t)e * F_DIM * H_DIM;
    const WT* w3e = w3 + (size_t)e * F_DIM * H_DIM;
    const WT* w2e = w2 + (size_t)e * H_DIM * F_DIM;

    const f32x8 vzero = {0.f,0.f,0.f,0.f,0.f,0.f,0.f,0.f};
    f32x8 acc[8];                                         // wave owns out cols [wave*128, +128)
#pragma unroll
    for (int t = 0; t < 8; ++t) acc[t] = vzero;

    for (int chunk = 0; chunk < F_DIM / FC; ++chunk) {
        // ---- phase 1: this wave's 16x16 tile of h = silu(x w1^T) * (x w3^T) ----
        const int f0 = chunk * FC + wave * 16;
        __builtin_prefetch(w1e + (size_t)(f0 + FC) * H_DIM, 0, 1);   // next chunk's rows
        __builtin_prefetch(w3e + (size_t)(f0 + FC) * H_DIM, 0, 1);
        f32x8 c1 = vzero, c3 = vzero;
#pragma unroll 4
        for (int k0 = 0; k0 < H_DIM; k0 += 32) {
            bf16x16 a  = load_a_lds(xs, H_DIM, k0, lane);
            bf16x16 b1 = load_b(w1e, (size_t)H_DIM, f0, k0, lane);
            c1 = wmma_bf16(a, b1, c1);
            bf16x16 b3 = load_b(w3e, (size_t)H_DIM, f0, k0, lane);
            c3 = wmma_bf16(a, b3, c3);
        }
        {   // silu + gate, store bf16 tile into hs (C-layout: m = r + 8*(lane>>4), n = lane&15)
            int n  = lane & 15;
            int mh = (lane >> 4) << 3;
            int col = wave * 16 + n;
#pragma unroll
            for (int r = 0; r < 8; ++r) {
                float t1  = c1[r];
                // silu(x) = x * sigmoid(x); use v_rcp_f32 instead of IEEE divide chain
                float sig = __builtin_amdgcn_rcpf(1.0f + __expf(-t1));
                float hv  = t1 * sig * c3[r];
                hs[(mh + r) * FC + col] = (__bf16)hv;
            }
        }
        __syncthreads();

        // ---- phase 2: acc[16 x 1024] += h_chunk @ w2[:, chunk]^T ----
#pragma unroll
        for (int ks = 0; ks < FC; ks += 32) {
            bf16x16 a = load_a_lds(hs, FC, ks, lane);
#pragma unroll
            for (int t = 0; t < 8; ++t) {
                int n0 = (wave * 8 + t) * 16;
                bf16x16 b = load_b(w2e, (size_t)F_DIM, n0, chunk * FC + ks, lane);
                acc[t] = wmma_bf16(a, b, acc[t]);
            }
        }
        __syncthreads();   // hs reused next chunk
    }

    // ---- epilogue: scale by routing weight, scatter-add (2 experts per token) ----
    {
        int n  = lane & 15;
        int mh = (lane >> 4) << 3;
#pragma unroll
        for (int t = 0; t < 8; ++t) {
            int col = (wave * 8 + t) * 16 + n;
#pragma unroll
            for (int r = 0; r < 8; ++r) {
                int m = mh + r;
                float v = acc[t][r] * twl[m];
                __hip_atomic_fetch_add(&out[(size_t)tokl[m] * H_DIM + col], v,
                                       __ATOMIC_RELAXED, __HIP_MEMORY_SCOPE_AGENT);
            }
        }
    }
}

// ---------------- host-side launch ----------------
extern "C" void kernel_launch(void* const* d_in, const int* in_sizes, int n_in,
                              void* d_out, int out_size, void* d_ws, size_t ws_size,
                              hipStream_t stream) {
    const float* x  = (const float*)d_in[0];   // [T, H]
    const float* gw = (const float*)d_in[1];   // [E, H]
    const float* w1 = (const float*)d_in[2];   // [E, F, H]
    const float* w3 = (const float*)d_in[3];   // [E, F, H]
    const float* w2 = (const float*)d_in[4];   // [E, H, F]

    float* out    = (float*)d_out;                       // [T, H]
    float* logits = out + (size_t)T_TOK * H_DIM;         // [T, E] (tuple output, concatenated)

    // workspace layout
    const size_t WELEM = (size_t)N_EXP * F_DIM * H_DIM;  // 33,554,432 elements per weight tensor
    char* ws = (char*)d_ws;
    size_t off = 0;
    int*    counts = (int*)(ws + off);   off += 256;
    int*    tixb   = (int*)(ws + off);   off += sizeof(int)   * (size_t)N_EXP * T_TOK;
    float*  twtb   = (float*)(ws + off); off += sizeof(float) * (size_t)N_EXP * T_TOK;
    __bf16* xb     = (__bf16*)(ws + off); off += sizeof(__bf16) * (size_t)T_TOK * H_DIM;
    off = (off + 255) & ~(size_t)255;
    __bf16* w1b    = (__bf16*)(ws + off); off += sizeof(__bf16) * WELEM;
    __bf16* w3b    = (__bf16*)(ws + off); off += sizeof(__bf16) * WELEM;
    __bf16* w2b    = (__bf16*)(ws + off); off += sizeof(__bf16) * WELEM;
    const bool use_bf16_w = (ws_size >= off);            // fixed per process -> deterministic

    hipLaunchKernelGGL(init_kernel, dim3(2048), dim3(256), 0, stream, out, counts);
    hipLaunchKernelGGL(cvt_x_kernel, dim3((T_TOK * H_DIM) / (256 * 4)), dim3(256), 0, stream, x, xb);
    hipLaunchKernelGGL(router_kernel, dim3(T_TOK / 8), dim3(256), 0, stream,
                       x, gw, logits, counts, tixb, twtb);

    if (use_bf16_w) {
        hipLaunchKernelGGL(cvt_w_kernel, dim3((unsigned)(WELEM / (256 * 4))), dim3(256), 0, stream,
                           w1, w1b, w3, w3b, w2, w2b);
        hipLaunchKernelGGL(moe_ffn_kernel<__bf16>, dim3(MAX_TILES, N_EXP), dim3(256), 0, stream,
                           w1b, w3b, w2b, xb, counts, tixb, twtb, out);
    } else {
        hipLaunchKernelGGL(moe_ffn_kernel<float>, dim3(MAX_TILES, N_EXP), dim3(256), 0, stream,
                           w1, w3, w2, xb, counts, tixb, twtb, out);
    }
}
